// Multi_Head_79594333930090
// MI455X (gfx1250) — matrix operand order repaired and checked by reading, not compiled, not run on hardware
//
#include <hip/hip_runtime.h>
#include <hip/hip_bf16.h>

typedef __attribute__((ext_vector_type(16))) _Float16 v16h;
typedef __attribute__((ext_vector_type(8)))  _Float16 v8h;
typedef __attribute__((ext_vector_type(8)))  float    v8f;

#define DEVFN static __device__ __forceinline__

static constexpr int kB  = 4;
static constexpr int kS  = 1024;
static constexpr int kD  = 1024;
static constexpr int kH  = 16;
static constexpr int kDv = 64;
static constexpr float kLnEps = 1e-5f;

// ---------------------------------------------------------------------------
// WMMA fragment helpers (wave32; layouts per CDNA5 ISA 7.12.2)
// ---------------------------------------------------------------------------

// A fragment 16x32 f16 from an f32 row-major matrix (convert on the fly).
DEVFN v16h load_a_from_f32(const float* __restrict__ X, long ld, int m0, int k0, int lane) {
  const int lr = lane & 15, sel = lane >> 4;
  const float* r0 = X + (long)(m0 + lr) * ld + k0 + sel * 8;
  v16h a;
#pragma unroll
  for (int j = 0; j < 8; ++j) a[j]     = (_Float16)r0[j];
#pragma unroll
  for (int j = 0; j < 8; ++j) a[j + 8] = (_Float16)r0[16 + j];
  return a;
}

// A fragment 16x32 from f16 source: two 16B loads per lane.
DEVFN v16h load_a_from_f16(const _Float16* __restrict__ X, long ld, int m0, int k0, int lane) {
  const int lr = lane & 15, sel = lane >> 4;
  const _Float16* r0 = X + (long)(m0 + lr) * ld + k0 + sel * 8;
  union { v16h v; v8h h[2]; } u;
  u.h[0] = *(const v8h*)r0;
  u.h[1] = *(const v8h*)(r0 + 16);
  return u.v;
}

// B fragment 32x16 f16: lane n%16 holds column n0+n; per-lane K in two 8-runs (k0+8*sel, k0+16+8*sel).
DEVFN v16h load_b_colmaj(const _Float16* __restrict__ Wt, long ld, int n0, int k0, int lane) {
  const int lr = lane & 15, sel = lane >> 4;
  const _Float16* p = Wt + (long)(n0 + lr) * ld + k0 + sel * 8;
  union { v16h v; v8h h[2]; } u;
  u.h[0] = *(const v8h*)p;
  u.h[1] = *(const v8h*)(p + 16);
  return u.v;
}

DEVFN v8f wmma16(v16h a, v16h b, v8f c) {
  // D = A x B + C : v_wmma_f32_16x16x32_f16
  return __builtin_amdgcn_wmma_f32_16x16x32_f16(false, a, false, b, (short)0, c, false, false);
}

// ---------------------------------------------------------------------------
// Kernels
// ---------------------------------------------------------------------------

__global__ __launch_bounds__(256)
void cvt_f32_to_f16_kernel(const float* __restrict__ src, _Float16* __restrict__ dst, int n) {
  int i = blockIdx.x * blockDim.x + threadIdx.x;
  if (i < n) dst[i] = (_Float16)src[i];
}

// QKV projection: X(4096x1024 f32) @ W^T(f16) + bias.
// z selects q/k/v. Q pre-scaled by 1/8, stored (b,h,S,dv) f16; K likewise;
// V stored transposed (b,h,dv,S) f16.
// Unroll-by-2 software pipeline: two buffer sets swap roles, no reg copies.
__global__ __launch_bounds__(256)
void qkv_proj_kernel(const float* __restrict__ q, const float* __restrict__ k,
                     const float* __restrict__ v,
                     const _Float16* __restrict__ Wq16, const _Float16* __restrict__ Wk16,
                     const _Float16* __restrict__ Wv16,
                     const float* __restrict__ bq, const float* __restrict__ bk,
                     const float* __restrict__ bv,
                     _Float16* __restrict__ Qh, _Float16* __restrict__ Kh,
                     _Float16* __restrict__ VhT) {
  const int lane = threadIdx.x & 31;
  const int wave = threadIdx.x >> 5;
  const int m0 = blockIdx.x * 128 + wave * 16;   // row in [0, B*S)
  const int n0 = blockIdx.y * 64;                // col in [0, D)
  const int p  = blockIdx.z;                     // 0=q 1=k 2=v

  const float*    X    = (p == 0) ? q    : (p == 1) ? k    : v;
  const _Float16* W    = (p == 0) ? Wq16 : (p == 1) ? Wk16 : Wv16;
  const float*    bias = (p == 0) ? bq   : (p == 1) ? bk   : bv;
  const float premul = (p == 0) ? 0.125f : 1.0f;   // 1/sqrt(dv)

  v8f acc[4] = {};
  v16h a0 = load_a_from_f32(X, kD, m0, 0, lane);
  v16h b0[4];
#pragma unroll
  for (int t = 0; t < 4; ++t) b0[t] = load_b_colmaj(W, kD, n0 + t * 16, 0, lane);

  for (int k0 = 0; k0 < kD; k0 += 64) {
    const int k1 = k0 + 32;                         // always in-bounds (64 | kD)
    const int k2 = (k0 + 64 < kD) ? k0 + 64 : k0;   // final trip: dead reload
    __builtin_prefetch(X + (long)(m0 + (lane & 15)) * kD + k0 + 128 + (lane >> 4) * 8, 0, 3);
    v16h a1 = load_a_from_f32(X, kD, m0, k1, lane);
    v16h b1[4];
#pragma unroll
    for (int t = 0; t < 4; ++t) b1[t] = load_b_colmaj(W, kD, n0 + t * 16, k1, lane);
#pragma unroll
    for (int t = 0; t < 4; ++t) acc[t] = wmma16(a0, b0[t], acc[t]);
    a0 = load_a_from_f32(X, kD, m0, k2, lane);
#pragma unroll
    for (int t = 0; t < 4; ++t) b0[t] = load_b_colmaj(W, kD, n0 + t * 16, k2, lane);
#pragma unroll
    for (int t = 0; t < 4; ++t) acc[t] = wmma16(a1, b1[t], acc[t]);
  }

  const int lr = lane & 15, sel = lane >> 4;
#pragma unroll
  for (int t = 0; t < 4; ++t) {
    const int n  = n0 + t * 16 + lr;
    const int hh = n >> 6, d = n & 63;
    const float bn = bias[n];
#pragma unroll
    for (int r = 0; r < 8; ++r) {
      const int row = m0 + sel * 8 + r;
      const int b_ = row >> 10, s = row & 1023;
      const float val = (acc[t][r] + bn) * premul;
      if (p == 2) {
        VhT[((long)(b_ * kH + hh) * kDv + d) * kS + s] = (_Float16)val;
      } else if (p == 0) {
        Qh[((long)(b_ * kH + hh) * kS + s) * kDv + d] = (_Float16)val;
      } else {
        Kh[((long)(b_ * kH + hh) * kS + s) * kDv + d] = (_Float16)val;
      }
    }
  }
}

// scores[bh, q, n] = sum_d Qh[bh,q,d] * Kh[bh,n,d]  (Q already scaled)
__global__ __launch_bounds__(256)
void scores_kernel(const _Float16* __restrict__ Qh, const _Float16* __restrict__ Kh,
                   float* __restrict__ att) {
  const int lane = threadIdx.x & 31;
  const int wave = threadIdx.x >> 5;
  const int q0 = blockIdx.x * 128 + wave * 16;
  const int n0 = blockIdx.y * 64;
  const int bh = blockIdx.z;
  const _Float16* Qb = Qh + (long)bh * kS * kDv;
  const _Float16* Kb = Kh + (long)bh * kS * kDv;
  float* Ab = att + (long)bh * kS * kS;

  // K=64: fully unrolled
  v16h a0 = load_a_from_f16(Qb, kDv, q0, 0, lane);
  v16h a1 = load_a_from_f16(Qb, kDv, q0, 32, lane);
  v8f acc[4] = {};
#pragma unroll
  for (int t = 0; t < 4; ++t) {
    v16h bf0 = load_b_colmaj(Kb, kDv, n0 + t * 16, 0, lane);
    v16h bf1 = load_b_colmaj(Kb, kDv, n0 + t * 16, 32, lane);
    acc[t] = wmma16(a0, bf0, acc[t]);
    acc[t] = wmma16(a1, bf1, acc[t]);
  }
  const int lr = lane & 15, sel = lane >> 4;
#pragma unroll
  for (int t = 0; t < 4; ++t)
#pragma unroll
    for (int r = 0; r < 8; ++r)
      Ab[(long)(q0 + sel * 8 + r) * kS + n0 + t * 16 + lr] = acc[t][r];
}

// In-place row softmax over 1024 columns; one block per row.
__global__ __launch_bounds__(256)
void softmax_kernel(float* __restrict__ att) {
  const long row = blockIdx.x;
  float* p = att + row * kS;
  __shared__ float red[256];
  const int t = threadIdx.x;
  float x0 = p[t], x1 = p[t + 256], x2 = p[t + 512], x3 = p[t + 768];
  float m = fmaxf(fmaxf(x0, x1), fmaxf(x2, x3));
  red[t] = m; __syncthreads();
  for (int off = 128; off > 0; off >>= 1) {
    if (t < off) red[t] = fmaxf(red[t], red[t + off]);
    __syncthreads();
  }
  m = red[0]; __syncthreads();
  x0 = __expf(x0 - m); x1 = __expf(x1 - m); x2 = __expf(x2 - m); x3 = __expf(x3 - m);
  red[t] = x0 + x1 + x2 + x3; __syncthreads();
  for (int off = 128; off > 0; off >>= 1) {
    if (t < off) red[t] += red[t + off];
    __syncthreads();
  }
  const float inv = 1.0f / red[0];
  p[t] = x0 * inv; p[t + 256] = x1 * inv; p[t + 512] = x2 * inv; p[t + 768] = x3 * inv;
}

// ctx[bh, q, d] = sum_k att[bh,q,k] * V[bh,k,d], using V^T layout for B frags.
// Stored as (b, s, h*dv) f16 so the out-projection A-fragments are contiguous.
__global__ __launch_bounds__(256)
void ctx_kernel(const float* __restrict__ att, const _Float16* __restrict__ VhT,
                _Float16* __restrict__ ctx16) {
  const int lane = threadIdx.x & 31;
  const int wave = threadIdx.x >> 5;
  const int q0 = blockIdx.x * 128 + wave * 16;
  const int bh = blockIdx.z;
  const int b_ = bh >> 4, hh = bh & 15;
  const float* Ab = att + (long)bh * kS * kS;
  const _Float16* Vb = VhT + (long)bh * kDv * kS;

  v8f acc[4] = {};
  v16h a0 = load_a_from_f32(Ab, kS, q0, 0, lane);
  v16h b0[4];
#pragma unroll
  for (int t = 0; t < 4; ++t) b0[t] = load_b_colmaj(Vb, kS, t * 16, 0, lane);

  for (int k0 = 0; k0 < kS; k0 += 64) {
    const int k1 = k0 + 32;
    const int k2 = (k0 + 64 < kS) ? k0 + 64 : k0;
    __builtin_prefetch(Ab + (long)(q0 + (lane & 15)) * kS + k0 + 128 + (lane >> 4) * 8, 0, 3);
    v16h a1 = load_a_from_f32(Ab, kS, q0, k1, lane);
    v16h b1[4];
#pragma unroll
    for (int t = 0; t < 4; ++t) b1[t] = load_b_colmaj(Vb, kS, t * 16, k1, lane);
#pragma unroll
    for (int t = 0; t < 4; ++t) acc[t] = wmma16(a0, b0[t], acc[t]);
    a0 = load_a_from_f32(Ab, kS, q0, k2, lane);
#pragma unroll
    for (int t = 0; t < 4; ++t) b0[t] = load_b_colmaj(Vb, kS, t * 16, k2, lane);
#pragma unroll
    for (int t = 0; t < 4; ++t) acc[t] = wmma16(a1, b1[t], acc[t]);
  }

  const int lr = lane & 15, sel = lane >> 4;
#pragma unroll
  for (int t = 0; t < 4; ++t)
#pragma unroll
    for (int r = 0; r < 8; ++r) {
      const int s = q0 + sel * 8 + r;
      ctx16[(long)(b_ * kS + s) * kD + hh * kDv + t * 16 + lr] = (_Float16)acc[t][r];
    }
}

// xres = q + ctx @ Wo^T + bo   (f32, 4096x1024)
__global__ __launch_bounds__(256)
void outproj_kernel(const _Float16* __restrict__ ctx16, const _Float16* __restrict__ Wo16,
                    const float* __restrict__ bo, const float* __restrict__ qin,
                    float* __restrict__ xres) {
  const int lane = threadIdx.x & 31;
  const int wave = threadIdx.x >> 5;
  const int m0 = blockIdx.x * 128 + wave * 16;
  const int n0 = blockIdx.y * 64;

  v8f acc[4] = {};
  v16h a0 = load_a_from_f16(ctx16, kD, m0, 0, lane);
  v16h b0[4];
#pragma unroll
  for (int t = 0; t < 4; ++t) b0[t] = load_b_colmaj(Wo16, kD, n0 + t * 16, 0, lane);

  for (int k0 = 0; k0 < kD; k0 += 64) {
    const int k1 = k0 + 32;
    const int k2 = (k0 + 64 < kD) ? k0 + 64 : k0;
    __builtin_prefetch(ctx16 + (long)(m0 + (lane & 15)) * kD + k0 + 128 + (lane >> 4) * 8, 0, 3);
    v16h a1 = load_a_from_f16(ctx16, kD, m0, k1, lane);
    v16h b1[4];
#pragma unroll
    for (int t = 0; t < 4; ++t) b1[t] = load_b_colmaj(Wo16, kD, n0 + t * 16, k1, lane);
#pragma unroll
    for (int t = 0; t < 4; ++t) acc[t] = wmma16(a0, b0[t], acc[t]);
    a0 = load_a_from_f16(ctx16, kD, m0, k2, lane);
#pragma unroll
    for (int t = 0; t < 4; ++t) b0[t] = load_b_colmaj(Wo16, kD, n0 + t * 16, k2, lane);
#pragma unroll
    for (int t = 0; t < 4; ++t) acc[t] = wmma16(a1, b1[t], acc[t]);
  }

  const int lr = lane & 15, sel = lane >> 4;
#pragma unroll
  for (int t = 0; t < 4; ++t) {
    const int n = n0 + t * 16 + lr;
    const float bn = bo[n];
#pragma unroll
    for (int r = 0; r < 8; ++r) {
      const long row = m0 + sel * 8 + r;
      xres[row * kD + n] = acc[t][r] + bn + qin[row * kD + n];
    }
  }
}

// Row LayerNorm over D=1024; one block per row.
__global__ __launch_bounds__(256)
void layernorm_kernel(const float* __restrict__ xres, const float* __restrict__ gamma,
                      const float* __restrict__ beta, float* __restrict__ y) {
  const long row = blockIdx.x;
  const float* p = xres + row * kD;
  float* o = y + row * kD;
  __shared__ float red[256];
  const int t = threadIdx.x;
  float x0 = p[t], x1 = p[t + 256], x2 = p[t + 512], x3 = p[t + 768];
  red[t] = x0 + x1 + x2 + x3; __syncthreads();
  for (int off = 128; off > 0; off >>= 1) {
    if (t < off) red[t] += red[t + off];
    __syncthreads();
  }
  const float mu = red[0] * (1.0f / kD); __syncthreads();
  const float d0 = x0 - mu, d1 = x1 - mu, d2 = x2 - mu, d3 = x3 - mu;
  red[t] = d0 * d0 + d1 * d1 + d2 * d2 + d3 * d3; __syncthreads();
  for (int off = 128; off > 0; off >>= 1) {
    if (t < off) red[t] += red[t + off];
    __syncthreads();
  }
  const float rstd = rsqrtf(red[0] * (1.0f / kD) + kLnEps);
  o[t]       = d0 * rstd * gamma[t]       + beta[t];
  o[t + 256] = d1 * rstd * gamma[t + 256] + beta[t + 256];
  o[t + 512] = d2 * rstd * gamma[t + 512] + beta[t + 512];
  o[t + 768] = d3 * rstd * gamma[t + 768] + beta[t + 768];
}

// ---------------------------------------------------------------------------
// Host launcher
// ---------------------------------------------------------------------------

extern "C" void kernel_launch(void* const* d_in, const int* in_sizes, int n_in,
                              void* d_out, int out_size, void* d_ws, size_t ws_size,
                              hipStream_t stream) {
  (void)in_sizes; (void)n_in; (void)out_size; (void)ws_size;

  const float* q     = (const float*)d_in[0];
  const float* k     = (const float*)d_in[1];
  const float* v     = (const float*)d_in[2];
  const float* Wq    = (const float*)d_in[3];
  const float* bq    = (const float*)d_in[4];
  const float* Wk    = (const float*)d_in[5];
  const float* bk    = (const float*)d_in[6];
  const float* Wv    = (const float*)d_in[7];
  const float* bv    = (const float*)d_in[8];
  const float* Wo    = (const float*)d_in[9];
  const float* bo    = (const float*)d_in[10];
  const float* gamma = (const float*)d_in[11];
  const float* beta  = (const float*)d_in[12];

  // d_out: y (B*S*D f32) followed by att (B*h*S*S f32)
  float* y   = (float*)d_out;
  float* att = (float*)d_out + (size_t)kB * kS * kD;

  // workspace layout (bytes)
  constexpr size_t MB = 1u << 20;
  char* ws = (char*)d_ws;
  _Float16* Wq16 = (_Float16*)(ws + 0 * MB);    // 2 MB
  _Float16* Wk16 = (_Float16*)(ws + 2 * MB);    // 2 MB
  _Float16* Wv16 = (_Float16*)(ws + 4 * MB);    // 2 MB
  _Float16* Wo16 = (_Float16*)(ws + 6 * MB);    // 2 MB
  _Float16* Qh   = (_Float16*)(ws + 8 * MB);    // 8 MB  (b,h,S,dv), pre-scaled
  _Float16* Kh   = (_Float16*)(ws + 16 * MB);   // 8 MB  (b,h,S,dv)
  _Float16* VhT  = (_Float16*)(ws + 24 * MB);   // 8 MB  (b,h,dv,S)
  _Float16* ctx16= (_Float16*)(ws + 32 * MB);   // 8 MB  (b,s,D)
  float*    xres = (float*)   (ws + 40 * MB);   // 16 MB (B*S, D)

  const int nW = kD * kD;
  const dim3 thr(256);
  cvt_f32_to_f16_kernel<<<dim3((nW + 255) / 256), thr, 0, stream>>>(Wq, Wq16, nW);
  cvt_f32_to_f16_kernel<<<dim3((nW + 255) / 256), thr, 0, stream>>>(Wk, Wk16, nW);
  cvt_f32_to_f16_kernel<<<dim3((nW + 255) / 256), thr, 0, stream>>>(Wv, Wv16, nW);
  cvt_f32_to_f16_kernel<<<dim3((nW + 255) / 256), thr, 0, stream>>>(Wo, Wo16, nW);

  qkv_proj_kernel<<<dim3(kB * kS / 128, kD / 64, 3), thr, 0, stream>>>(
      q, k, v, Wq16, Wk16, Wv16, bq, bk, bv, Qh, Kh, VhT);

  scores_kernel<<<dim3(kS / 128, kS / 64, kB * kH), thr, 0, stream>>>(Qh, Kh, att);

  softmax_kernel<<<dim3(kB * kH * kS), thr, 0, stream>>>(att);

  ctx_kernel<<<dim3(kS / 128, 1, kB * kH), thr, 0, stream>>>(att, VhT, ctx16);

  outproj_kernel<<<dim3(kB * kS / 128, kD / 64, 1), thr, 0, stream>>>(
      ctx16, Wo16, bo, q, xres);

  layernorm_kernel<<<dim3(kB * kS), thr, 0, stream>>>(xres, gamma, beta, y);
}
